// AttentionBlock_13365938225820
// MI455X (gfx1250) — matrix operand order repaired and checked by reading, not compiled, not run on hardware
//
#include <hip/hip_runtime.h>

// ---------------------------------------------------------------------------
// AttentionBlock for MI455X (gfx1250): bf16 WMMA everywhere, f32 accumulate.
// Pipeline: prep_w -> gn_stats -> gn_apply -> qkv_gemm -> fused_attn -> proj
// ---------------------------------------------------------------------------

#define CCH   512
#define NTOK  1024
#define NB    16
#define NGRP  8
#define GSIZE 65536                   // (C/NGRP)*NTOK elements per (b,g)
#define SCALE 0.04419417382415922f    // 512^-0.5 (folded into q at QKV store)

typedef __attribute__((ext_vector_type(16))) __bf16 v16bf;
typedef __attribute__((ext_vector_type(8)))  __bf16 v8bf;
typedef __attribute__((ext_vector_type(8)))  float  v8f;
typedef __attribute__((ext_vector_type(8)))  unsigned short v8us;
typedef __attribute__((ext_vector_type(4)))  unsigned short v4us;

// f32 -> bf16 (round to nearest even)
__device__ __forceinline__ unsigned short f2bf(float f) {
    union { float f; unsigned u; } v; v.f = f;
    unsigned r = v.u + 0x7FFFu + ((v.u >> 16) & 1u);
    return (unsigned short)(r >> 16);
}

// A-fragment: row = per-lane row pointer (ushort units), K-chunk base k0.
// ISA layout: lanes 0-15 hold K = k0+{0..7, 16..23}; lanes 16-31 hold +8.
__device__ __forceinline__ v16bf load_a(const unsigned short* row, int k0, int lane) {
    int off = (lane & 16) ? 8 : 0;
    v8bf lo = *(const v8bf*)(row + k0 + off);
    v8bf hi = *(const v8bf*)(row + k0 + 16 + off);
    return __builtin_shufflevector(lo, hi, 0,1,2,3,4,5,6,7,8,9,10,11,12,13,14,15);
}
// B-fragment: lane's column source row; lanes 0-15: K=k0+{0..7, 16..23}, lanes 16-31: +8.
__device__ __forceinline__ v16bf load_b(const unsigned short* row, int k0, int lane) {
    int off = (lane & 16) ? 8 : 0;
    v8bf lo = *(const v8bf*)(row + k0 + off);
    v8bf hi = *(const v8bf*)(row + k0 + 16 + off);
    return __builtin_shufflevector(lo, hi, 0,1,2,3,4,5,6,7,8,9,10,11,12,13,14,15);
}
__device__ __forceinline__ v8f wmma_bf(v16bf a, v16bf b, v8f c) {
    return __builtin_amdgcn_wmma_f32_16x16x32_bf16(false, a, false, b, (short)0, c, false, false);
}

// ---------------- weight conversion f32 -> bf16 ----------------
__global__ void prep_w_kernel(const float* __restrict__ wq, const float* __restrict__ wp,
                              unsigned short* __restrict__ wq_bf, unsigned short* __restrict__ wp_bf) {
    int tid = blockIdx.x * 256 + threadIdx.x;
    if (tid < 3 * CCH * CCH) wq_bf[tid] = f2bf(wq[tid]);
    else {
        int t = tid - 3 * CCH * CCH;
        if (t < CCH * CCH) wp_bf[t] = f2bf(wp[t]);
    }
}

// ---------------- GroupNorm stats: one block per (b,g) ----------------
__global__ void gn_stats_kernel(const float* __restrict__ x, float* __restrict__ stats) {
    int bg = blockIdx.x;
    const float* base = x + (size_t)bg * GSIZE;   // group channels are contiguous
    int t = threadIdx.x;
    float s = 0.f, ss = 0.f;
    for (int i = 0; i < GSIZE / 256; ++i) { float v = base[t + 256 * i]; s += v; ss += v * v; }
    __shared__ float sa[256], sb[256];
    sa[t] = s; sb[t] = ss; __syncthreads();
    for (int w = 128; w > 0; w >>= 1) {
        if (t < w) { sa[t] += sa[t + w]; sb[t] += sb[t + w]; }
        __syncthreads();
    }
    if (t == 0) {
        float mean = sa[0] * (1.f / GSIZE);
        float var  = sb[0] * (1.f / GSIZE) - mean * mean;
        stats[2 * bg] = mean; stats[2 * bg + 1] = rsqrtf(var + 1e-5f);
    }
}

// ---------------- GroupNorm apply: write hT token-major bf16 ----------------
__global__ void gn_apply_kernel(const float* __restrict__ x, const float* __restrict__ stats,
                                const float* __restrict__ w, const float* __restrict__ b,
                                unsigned short* __restrict__ hT) {
    int tid = blockIdx.x * 256 + threadIdx.x;     // tid = (b*N + n)*C + c
    int c = tid & (CCH - 1);
    int n = (tid >> 9) & (NTOK - 1);
    int bb = tid >> 19;
    float xv = x[(((size_t)bb * CCH + c) << 10) + n];
    int bg = bb * NGRP + (c >> 6);
    float hn = (xv - stats[2 * bg]) * stats[2 * bg + 1] * w[c] + b[c];
    hT[tid] = f2bf(hn);
}

// ---------------- QKV GEMM: one wave per 32x32 macro-tile of [3C x N] --------
__global__ void qkv_kernel(const unsigned short* __restrict__ hT,
                           const unsigned short* __restrict__ wq_bf,
                           const float* __restrict__ qb,
                           unsigned short* __restrict__ qT, unsigned short* __restrict__ kT,
                           unsigned short* __restrict__ vbuf) {
    int lane = threadIdx.x & 31, wave = threadIdx.x >> 5;
    int gid = blockIdx.x * 4 + wave;              // 16 * 48 * 32 waves
    int bb = gid / (48 * 32); int rem = gid % (48 * 32);
    int o0 = (rem / 32) * 32, n0 = (rem % 32) * 32;
    const unsigned short* ar0 = wq_bf + (size_t)(o0 + (lane & 15)) * CCH;
    const unsigned short* ar1 = ar0 + 16 * CCH;
    const unsigned short* br0 = hT + ((size_t)bb * NTOK + n0 + (lane & 15)) * CCH;
    const unsigned short* br1 = br0 + 16 * CCH;
    v8f a00 = {}, a01 = {}, a10 = {}, a11 = {};
#pragma unroll
    for (int kc = 0; kc < 16; ++kc) {
        int k0 = kc * 32;
        v16bf fa0 = load_a(ar0, k0, lane), fa1 = load_a(ar1, k0, lane);
        v16bf fb0 = load_b(br0, k0, lane), fb1 = load_b(br1, k0, lane);
        a00 = wmma_bf(fa0, fb0, a00); a01 = wmma_bf(fa0, fb1, a01);
        a10 = wmma_bf(fa1, fb0, a10); a11 = wmma_bf(fa1, fb1, a11);
    }
    int half8 = (lane & 16) ? 8 : 0;
    v8f accs[4] = { a00, a01, a10, a11 };
#pragma unroll
    for (int s = 0; s < 4; ++s) {
        int oi = (s >> 1) * 16, ni = (s & 1) * 16;
        int obase = o0 + oi + half8;
        int nrow  = n0 + ni + (lane & 15);
        if (o0 < 2 * CCH) {   // q or k: transposed pack-store (8 contiguous o per lane)
            float mul = (o0 < CCH) ? SCALE : 1.0f;   // fold softmax scale into q
            v8us pk;
#pragma unroll
            for (int r = 0; r < 8; ++r) pk[r] = f2bf((accs[s][r] + qb[obase + r]) * mul);
            unsigned short* dst = (o0 < CCH) ? qT : kT;
            int oo = (o0 < CCH) ? obase : (obase - CCH);
            *(v8us*)(dst + ((size_t)bb * NTOK + nrow) * CCH + oo) = pk;
        } else {              // v: channel-major
#pragma unroll
            for (int r = 0; r < 8; ++r) {
                int c = obase - 2 * CCH + r;
                vbuf[((size_t)bb * CCH + c) * NTOK + nrow] = f2bf(accs[s][r] + qb[obase + r]);
            }
        }
    }
}

// ---------------- Fused attention: one block (4 waves) per (b, 16-row tile) ----
__global__ void attn_kernel(const unsigned short* __restrict__ qT,
                            const unsigned short* __restrict__ kT,
                            const unsigned short* __restrict__ vbuf,
                            unsigned short* __restrict__ outT) {
    int lane = threadIdx.x & 31, wave = threadIdx.x >> 5;
    int bb = blockIdx.x >> 6;
    int i0 = (blockIdx.x & 63) * 16;
    int half8 = (lane & 16) ? 8 : 0;

    __shared__ float S_part[4][16 * 32];          // per-wave K-slice partial logits
    __shared__ unsigned short P_lds[16 * 32];     // exp(S) in bf16 (A-fragment source)
    __shared__ float rpart[8][16];
    __shared__ float dinv_s[16];
    __shared__ unsigned short stage[4][16 * 128];

    // preload q A-fragments for this wave's 128-channel K-slice
    const unsigned short* qrow = qT + ((size_t)bb * NTOK + i0 + (lane & 15)) * CCH;
    v16bf qa[4];
#pragma unroll
    for (int t = 0; t < 4; ++t) qa[t] = load_a(qrow, wave * 128 + t * 32, lane);

    v8f O[8];
#pragma unroll
    for (int i = 0; i < 8; ++i) O[i] = (v8f){};
    float rsum = 0.f;
    int row16 = lane & 15;
    int cg = (wave << 1) | (lane >> 4);           // 0..7

    for (int jt = 0; jt < 32; ++jt) {
        int j0 = jt * 32;
        // logits: S[16][32] partial over c in [wave*128, wave*128+128)
        // (each wave fully overwrites its own S_part region -> no zeroing needed)
#pragma unroll
        for (int sub = 0; sub < 2; ++sub) {
            const unsigned short* krow = kT + ((size_t)bb * NTOK + j0 + sub * 16 + (lane & 15)) * CCH;
            v8f s = {};
#pragma unroll
            for (int t = 0; t < 4; ++t)
                s = wmma_bf(qa[t], load_b(krow, wave * 128 + t * 32, lane), s);
#pragma unroll
            for (int r = 0; r < 8; ++r)
                S_part[wave][(half8 + r) * 32 + sub * 16 + (lane & 15)] = s[r];
        }
        __syncthreads();
        // reduce partials, exp, pack bf16 P, accumulate row sums
#pragma unroll
        for (int t = 0; t < 4; ++t) {
            int col = cg + 8 * t;
            int idx = row16 * 32 + col;
            float sv = S_part[0][idx] + S_part[1][idx] + S_part[2][idx] + S_part[3][idx];
            float e = __expf(sv);                 // scale already folded into q
            rsum += e;
            P_lds[idx] = f2bf(e);
        }
        __syncthreads();
        // P @ V^T for this wave's 128-channel output slice
        v16bf pa = load_a(P_lds + (size_t)row16 * 32, 0, lane);
#pragma unroll
        for (int ct = 0; ct < 8; ++ct) {
            int c = wave * 128 + ct * 16 + (lane & 15);
            const unsigned short* vrow = vbuf + ((size_t)bb * CCH + c) * NTOK;
            O[ct] = wmma_bf(pa, load_b(vrow, j0, lane), O[ct]);
        }
        // next iteration's S_part writes are safe: every thread reaching them has
        // passed the post-exp barrier, after which no thread reads S_part again.
    }

    // row-sum reduction -> 1/d
    rpart[cg][row16] = rsum;
    __syncthreads();
    if (threadIdx.x < 16) {
        float d = 0.f;
#pragma unroll
        for (int g = 0; g < 8; ++g) d += rpart[g][threadIdx.x];
        dinv_s[threadIdx.x] = 1.f / d;
    }
    __syncthreads();
    // normalize, transpose via LDS, coalesced bf16 store (token-major outT)
#pragma unroll
    for (int ct = 0; ct < 8; ++ct)
#pragma unroll
        for (int r = 0; r < 8; ++r) {
            int row = half8 + r;
            stage[wave][row * 128 + ct * 16 + (lane & 15)] = f2bf(O[ct][r] * dinv_s[row]);
        }
    __syncthreads();
    for (int t = 0; t < 16; ++t) {
        v4us val = *(v4us*)&stage[wave][t * 128 + lane * 4];
        *(v4us*)(outT + ((size_t)bb * NTOK + i0 + t) * CCH + wave * 128 + lane * 4) = val;
    }
}

// ---------------- Projection + bias + residual (32x32 macro-tiles) ----------
__global__ void proj_kernel(const unsigned short* __restrict__ outT,
                            const unsigned short* __restrict__ wp_bf,
                            const float* __restrict__ pb,
                            const float* __restrict__ x,
                            float* __restrict__ y) {
    int lane = threadIdx.x & 31, wave = threadIdx.x >> 5;
    int gid = blockIdx.x * 4 + wave;              // 16 * 16 * 32 waves
    int bb = gid / (16 * 32); int rem = gid % (16 * 32);
    int o0 = (rem / 32) * 32, n0 = (rem % 32) * 32;
    const unsigned short* ar0 = wp_bf + (size_t)(o0 + (lane & 15)) * CCH;
    const unsigned short* ar1 = ar0 + 16 * CCH;
    const unsigned short* br0 = outT + ((size_t)bb * NTOK + n0 + (lane & 15)) * CCH;
    const unsigned short* br1 = br0 + 16 * CCH;
    v8f a00 = {}, a01 = {}, a10 = {}, a11 = {};
#pragma unroll
    for (int kc = 0; kc < 16; ++kc) {
        int k0 = kc * 32;
        v16bf fa0 = load_a(ar0, k0, lane), fa1 = load_a(ar1, k0, lane);
        v16bf fb0 = load_b(br0, k0, lane), fb1 = load_b(br1, k0, lane);
        a00 = wmma_bf(fa0, fb0, a00); a01 = wmma_bf(fa0, fb1, a01);
        a10 = wmma_bf(fa1, fb0, a10); a11 = wmma_bf(fa1, fb1, a11);
    }
    int half8 = (lane & 16) ? 8 : 0;
    v8f accs[4] = { a00, a01, a10, a11 };
#pragma unroll
    for (int s = 0; s < 4; ++s) {
        int oi = (s >> 1) * 16, ni = (s & 1) * 16;
#pragma unroll
        for (int r = 0; r < 8; ++r) {
            int o = o0 + oi + half8 + r;
            size_t idx = ((size_t)bb * CCH + o) * NTOK + n0 + ni + (lane & 15);
            y[idx] = x[idx] + accs[s][r] + pb[o];
        }
    }
}

// ---------------- launch ----------------
extern "C" void kernel_launch(void* const* d_in, const int* in_sizes, int n_in,
                              void* d_out, int out_size, void* d_ws, size_t ws_size,
                              hipStream_t stream) {
    const float* x      = (const float*)d_in[0];
    const float* norm_w = (const float*)d_in[1];
    const float* norm_b = (const float*)d_in[2];
    const float* qkv_w  = (const float*)d_in[3];
    const float* qkv_b  = (const float*)d_in[4];
    const float* proj_w = (const float*)d_in[5];
    const float* proj_b = (const float*)d_in[6];
    float* out = (float*)d_out;
    char* ws = (char*)d_ws;

    const size_t ACT = (size_t)NB * NTOK * CCH * 2;   // 16 MB bf16 activation buffer
    unsigned short* wq_bf = (unsigned short*)(ws);                   // 1.5 MB
    unsigned short* wp_bf = (unsigned short*)(ws + 0x180000);        // 0.5 MB
    float*          stats = (float*)(ws + 0x200000);                 // 1 KB
    unsigned short* hT    = (unsigned short*)(ws + 0x210000);
    unsigned short* qT    = (unsigned short*)(ws + 0x210000 + ACT);
    unsigned short* kT    = (unsigned short*)(ws + 0x210000 + 2 * ACT);
    unsigned short* vbuf  = (unsigned short*)(ws + 0x210000 + 3 * ACT);
    unsigned short* outT  = (unsigned short*)(ws + 0x210000 + 4 * ACT);

    prep_w_kernel<<<4096, 256, 0, stream>>>(qkv_w, proj_w, wq_bf, wp_bf);
    gn_stats_kernel<<<NB * NGRP, 256, 0, stream>>>(x, stats);
    gn_apply_kernel<<<(NB * CCH * NTOK) / 256, 256, 0, stream>>>(x, stats, norm_w, norm_b, hT);
    qkv_kernel<<<(NB * 48 * 32) / 4, 128, 0, stream>>>(hT, wq_bf, qkv_b, qT, kT, vbuf);
    attn_kernel<<<NB * 64, 128, 0, stream>>>(qT, kT, vbuf, outT);
    proj_kernel<<<(NB * 16 * 32) / 4, 128, 0, stream>>>(outT, wp_bf, proj_b, x, out);
}
